// Head_68710886801526
// MI455X (gfx1250) — matrix operand order repaired and checked by reading, not compiled, not run on hardware
//
#include <hip/hip_runtime.h>
#include <hip/hip_bf16.h>

#define BB 64
#define TT 512
#define EE 512
#define HH 64

typedef __attribute__((ext_vector_type(16))) __bf16 v16bf;
typedef __attribute__((ext_vector_type(8)))  __bf16 v8bf;
typedef __attribute__((ext_vector_type(8)))  float  v8f;
typedef __attribute__((ext_vector_type(4))) unsigned int u32x4;
typedef __attribute__((ext_vector_type(8))) int          i32x8;
typedef __attribute__((ext_vector_type(4))) int          i32x4;

#if __has_builtin(__builtin_amdgcn_tensor_load_to_lds)
#define HAVE_TDM 1
#else
#define HAVE_TDM 0
#endif

static __device__ __forceinline__ v8f wmma_bf16(v16bf a, v16bf b, v8f c) {
  // D = A(16x32 bf16) * B(32x16 bf16) + C(16x16 f32)
  return __builtin_amdgcn_wmma_f32_16x16x32_bf16(false, a, false, b, (short)0, c,
                                                 false, false);
}

#if HAVE_TDM
// TDM: load a 2-D tile (tile_dim0=32 contiguous bf16, tile_dim1=192 rows,
// row stride 512 elements) from global into LDS (rows compacted to 32 elems).
static __device__ __forceinline__ void tdm_load_w(const __bf16* gsrc,
                                                  unsigned lds_byte_addr) {
  unsigned long long ga = (unsigned long long)(size_t)gsrc;
  u32x4 g0;
  g0[0] = 1u;                                   // count=1 (valid user D#)
  g0[1] = lds_byte_addr;                        // lds_addr [63:32]
  g0[2] = (unsigned)(ga & 0xFFFFFFFFu);         // global_addr lo
  g0[3] = (unsigned)((ga >> 32) & 0x01FFFFFFu)  // global_addr [56:32]
          | (2u << 30);                         // type=2 ("image")
  i32x8 g1;
  g1[0] = 0x00010000;          // workgroup_mask=0, data_size=1 (2 bytes)
  g1[1] = (int)(512u << 16);   // tensor_dim0 = 512 (lo16 @ bits 63:48)
  g1[2] = (int)(192u << 16);   // tensor_dim0 hi=0 | tensor_dim1 = 192
  g1[3] = (int)(32u  << 16);   // tensor_dim1 hi=0 | tile_dim0 = 32
  g1[4] = 192;                 // tile_dim1 = 192, tile_dim2 = 0
  g1[5] = 512;                 // tensor_dim0_stride lo32 = 512
  g1[6] = 0;                   // stride0 hi | stride1 lo
  g1[7] = 0;                   // stride1 hi
  i32x4 z4 = (i32x4){0, 0, 0, 0};
#if defined(__clang_major__) && (__clang_major__ >= 23)
  i32x8 z8 = (i32x8){0, 0, 0, 0, 0, 0, 0, 0};
  __builtin_amdgcn_tensor_load_to_lds(g0, g1, z4, z4, z8, 0);
#else
  __builtin_amdgcn_tensor_load_to_lds(g0, g1, z4, z4, 0);
#endif
}
#else
// Fallback: cooperative synchronous copy of the 192x32 slice.
static __device__ __forceinline__ void coop_load_w(const __bf16* gsrc,
                                                   __bf16* dst, int tid) {
  for (int idx = tid; idx < 192 * 4; idx += 256) {   // 8 bf16 per chunk
    int row = idx >> 2, c8 = (idx & 3) * 8;
    *(v8bf*)(dst + row * 32 + c8) = *(const v8bf*)(gsrc + row * 512 + c8);
  }
}
#endif

// ---------------------------------------------------------------------------
// Kernel 0: transpose + convert Wq/Wk/Wv (E x H f32) -> (H x E bf16),
// stored contiguously (WTq | WTk | WTv) so one TDM tile covers all three.
// ---------------------------------------------------------------------------
__global__ void wt_kernel(const float* __restrict__ Wk, const float* __restrict__ Wq,
                          const float* __restrict__ Wv,
                          __bf16* __restrict__ WTq, __bf16* __restrict__ WTk,
                          __bf16* __restrict__ WTv) {
  int idx = blockIdx.x * blockDim.x + threadIdx.x;   // over H*E
  if (idx >= HH * EE) return;
  int h = idx / EE;
  int e = idx - h * EE;
  WTq[idx] = (__bf16)Wq[e * HH + h];
  WTk[idx] = (__bf16)Wk[e * HH + h];
  WTv[idx] = (__bf16)Wv[e * HH + h];
}

// ---------------------------------------------------------------------------
// Kernel 1: Q = xWq, K = xWk, V = xWv with W slices staged in LDS by the TDM
// (double-buffered, overlapped with WMMA). Grid: B*4 blocks x 256 threads.
// ---------------------------------------------------------------------------
__global__ __launch_bounds__(256) void qkv_proj_kernel(
    const float* __restrict__ x, const __bf16* __restrict__ WT,
    __bf16* __restrict__ Qb, __bf16* __restrict__ Kb, __bf16* __restrict__ VT) {
  __shared__ __align__(128) __bf16 wbuf[2][192 * 32];  // 2 x 12KB

  const int lane = threadIdx.x & 31;
  const int w    = threadIdx.x >> 5;      // wave in block: 0..7
  const int n    = lane & 15;             // A-row / B-col within fragment
  const int kh   = lane >> 4;             // K-half selector
  const int b    = blockIdx.x >> 2;       // batch
  const int rblk = blockIdx.x & 3;        // 128-row block within T
  const int r0   = rblk * 128 + w * 16;   // wave's first row

  const float* xrow = x + ((size_t)b * TT + r0 + n) * EE;

  v8f acc[12];                            // [proj*4 + ct]
#pragma unroll
  for (int i = 0; i < 12; ++i) acc[i] = (v8f){0.f,0.f,0.f,0.f,0.f,0.f,0.f,0.f};

  // prologue: fill buffer 0
#if HAVE_TDM
  if (w == 0) {
    tdm_load_w(WT, (unsigned)(size_t)&wbuf[0][0]);
    __builtin_amdgcn_s_wait_tensorcnt(0);
  }
  __syncthreads();
#else
  coop_load_w(WT, &wbuf[0][0], threadIdx.x);
  __syncthreads();
#endif

  for (int ks = 0; ks < EE / 32; ++ks) {
    const int cur = ks & 1;
#if HAVE_TDM
    if (w == 0 && ks + 1 < EE / 32)       // async prefetch next slice
      tdm_load_w(WT + (ks + 1) * 32, (unsigned)(size_t)&wbuf[cur ^ 1][0]);
#endif

    // A fragment: 16x32 of x, f32 -> bf16 on load.
    const int e0 = ks * 32 + kh * 16;
    v16bf a;
#pragma unroll
    for (int j = 0; j < 8; ++j) {
      a[j]     = (__bf16)xrow[e0 + j];
      a[j + 8] = (__bf16)xrow[e0 + 8 + j];
    }

    // 12 WMMAs with B fragments from LDS (row = p*64 + ct*16 + n).
#pragma unroll
    for (int p = 0; p < 3; ++p) {
#pragma unroll
      for (int ct = 0; ct < 4; ++ct) {
        const __bf16* brow = &wbuf[cur][(p * 64 + ct * 16 + n) * 32 + kh * 16];
        v16bf bf = *(const v16bf*)brow;
        acc[p * 4 + ct] = wmma_bf16(a, bf, acc[p * 4 + ct]);
      }
    }

    __syncthreads();                      // all waves done reading wbuf[cur]
#if HAVE_TDM
    if (w == 0 && ks + 1 < EE / 32)
      __builtin_amdgcn_s_wait_tensorcnt(0);
    __syncthreads();                      // wbuf[cur^1] now valid for everyone
#else
    if (ks + 1 < EE / 32)
      coop_load_w(WT + (ks + 1) * 32, &wbuf[cur ^ 1][0], threadIdx.x);
    __syncthreads();
#endif
  }

  // C layout: lane holds column n, rows m = i + 8*kh.
#pragma unroll
  for (int ct = 0; ct < 4; ++ct) {
#pragma unroll
    for (int i = 0; i < 8; ++i) {
      const int t = r0 + i + 8 * kh;
      Qb[((size_t)b * TT + t) * HH + ct * 16 + n] = (__bf16)acc[0 * 4 + ct][i];
      Kb[((size_t)b * TT + t) * HH + ct * 16 + n] = (__bf16)acc[1 * 4 + ct][i];
    }
    // V stored transposed: VT[b][h][t]; rows m contiguous in t -> 16B store.
    v8bf pv;
#pragma unroll
    for (int i = 0; i < 8; ++i) pv[i] = (__bf16)acc[2 * 4 + ct][i];
    *(v8bf*)(VT + ((size_t)b * HH + ct * 16 + n) * TT + r0 + 8 * kh) = pv;
  }
}

// ---------------------------------------------------------------------------
// Kernel 2: flash attention. One wave per (batch, 16-query tile).
// ---------------------------------------------------------------------------
__global__ __launch_bounds__(256) void attn_kernel(
    const __bf16* __restrict__ Qb, const __bf16* __restrict__ Kb,
    const __bf16* __restrict__ VT, const int* __restrict__ amask,
    float* __restrict__ out) {
  __shared__ __align__(64) __bf16 plds[8][16 * 32];   // per-wave P staging

  const int lane = threadIdx.x & 31;
  const int w    = threadIdx.x >> 5;
  const int n    = lane & 15;
  const int kh   = lane >> 4;

  const int gw = blockIdx.x * 8 + w;      // global wave id, 0..2047
  const int b  = gw >> 5;                 // 32 query tiles per batch
  const int qt = gw & 31;

  // Q fragments: A-operand 16x32, two K-steps covering H=64.
  const __bf16* qrow = Qb + ((size_t)b * TT + qt * 16 + n) * HH;
  v16bf qf[2];
#pragma unroll
  for (int s = 0; s < 2; ++s) {
    v8bf c0 = *(const v8bf*)(qrow + s * 32 + kh * 16);
    v8bf c1 = *(const v8bf*)(qrow + s * 32 + kh * 16 + 8);
#pragma unroll
    for (int j = 0; j < 8; ++j) { qf[s][j] = c0[j]; qf[s][j + 8] = c1[j]; }
  }

  v8f o[4];
#pragma unroll
  for (int ct = 0; ct < 4; ++ct) o[ct] = (v8f){0.f,0.f,0.f,0.f,0.f,0.f,0.f,0.f};
  float mrow[8], lrow[8];
#pragma unroll
  for (int i = 0; i < 8; ++i) { mrow[i] = -1e30f; lrow[i] = 0.f; }

  const float scale = 0.125f;             // H^-0.5

  const int nks = ((qt * 16 + 15) >> 5) + 1;   // causal: skip fully-masked tiles
  for (int ks = 0; ks < nks; ++ks) {
    const int s0 = ks * 32;

    // ---- scores: two 16x16 tiles covering 32 keys, K-dim = H = 64 ----
    v8f sc[2];
#pragma unroll
    for (int t2 = 0; t2 < 2; ++t2) {
      const __bf16* krow = Kb + ((size_t)b * TT + s0 + t2 * 16 + n) * HH;
      v8f s = (v8f){0.f,0.f,0.f,0.f,0.f,0.f,0.f,0.f};
      s = wmma_bf16(qf[0], *(const v16bf*)(krow + 0  + kh * 16), s);
      s = wmma_bf16(qf[1], *(const v16bf*)(krow + 32 + kh * 16), s);
      sc[t2] = s;
    }
    if (ks + 1 < nks)   // warm L2/WGP$ for next key tile
      __builtin_prefetch(Kb + ((size_t)b * TT + s0 + 32 + n) * HH, 0, 0);

    // ---- scale + causal + padding mask ----
    const int key0 = s0 + n, key1 = s0 + 16 + n;
    const int pm0 = amask[b * TT + key0];
    const int pm1 = amask[b * TT + key1];
#pragma unroll
    for (int i = 0; i < 8; ++i) {
      const int tq = qt * 16 + i + 8 * kh;
      float v0 = sc[0][i] * scale;
      float v1 = sc[1][i] * scale;
      if (key0 > tq || pm0 == 0) v0 = -1e30f;
      if (key1 > tq || pm1 == 0) v1 = -1e30f;
      sc[0][i] = v0; sc[1][i] = v1;
    }

    // ---- online softmax (rows live in 16-lane halves of C fragment) ----
#pragma unroll
    for (int i = 0; i < 8; ++i) {
      float rm = fmaxf(sc[0][i], sc[1][i]);
      rm = fmaxf(rm, __shfl_xor(rm, 1, 16));
      rm = fmaxf(rm, __shfl_xor(rm, 2, 16));
      rm = fmaxf(rm, __shfl_xor(rm, 4, 16));
      rm = fmaxf(rm, __shfl_xor(rm, 8, 16));
      const float mnew  = fmaxf(mrow[i], rm);
      const float alpha = __expf(mrow[i] - mnew);
      const float p0 = __expf(sc[0][i] - mnew);
      const float p1 = __expf(sc[1][i] - mnew);
      float rs = p0 + p1;
      rs += __shfl_xor(rs, 1, 16);
      rs += __shfl_xor(rs, 2, 16);
      rs += __shfl_xor(rs, 4, 16);
      rs += __shfl_xor(rs, 8, 16);
      lrow[i] = lrow[i] * alpha + rs;
      mrow[i] = mnew;
#pragma unroll
      for (int ct = 0; ct < 4; ++ct) o[ct][i] *= alpha;
      // stash P (C layout) to LDS for A-layout reload
      plds[w][(i + 8 * kh) * 32 + n]      = (__bf16)p0;
      plds[w][(i + 8 * kh) * 32 + 16 + n] = (__bf16)p1;
    }

    // DS ops of a wave are in-order; wait for stores, then reload as A fragment.
    asm volatile("s_wait_dscnt 0" ::: "memory");
    v16bf pa;
    {
      v8bf c0 = *(const v8bf*)&plds[w][n * 32 + kh * 16];
      v8bf c1 = *(const v8bf*)&plds[w][n * 32 + kh * 16 + 8];
#pragma unroll
      for (int j = 0; j < 8; ++j) { pa[j] = c0[j]; pa[j + 8] = c1[j]; }
    }
    asm volatile("" ::: "memory");  // keep next iter's DS stores after these loads

    // ---- O += P @ V : B fragment from transposed V, contiguous in s ----
#pragma unroll
    for (int ct = 0; ct < 4; ++ct) {
      const __bf16* vrow = VT + ((size_t)b * HH + ct * 16 + n) * TT + s0 + kh * 16;
      o[ct] = wmma_bf16(pa, *(const v16bf*)vrow, o[ct]);
    }
  }

  // ---- normalize and store f32 output (B,T,H) ----
#pragma unroll
  for (int i = 0; i < 8; ++i) {
    const float inv = lrow[i] > 0.f ? 1.0f / lrow[i] : 0.f;
    const int t = qt * 16 + i + 8 * kh;
#pragma unroll
    for (int ct = 0; ct < 4; ++ct)
      out[((size_t)b * TT + t) * HH + ct * 16 + n] = o[ct][i] * inv;
  }
}

// ---------------------------------------------------------------------------
extern "C" void kernel_launch(void* const* d_in, const int* in_sizes, int n_in,
                              void* d_out, int out_size, void* d_ws, size_t ws_size,
                              hipStream_t stream) {
  (void)in_sizes; (void)n_in; (void)out_size; (void)ws_size;
  const float* x   = (const float*)d_in[0];
  const int* amask = (const int*)d_in[1];
  const float* Wk  = (const float*)d_in[2];   // dict order: Wk, Wq, Wv
  const float* Wq  = (const float*)d_in[3];
  const float* Wv  = (const float*)d_in[4];
  float* out = (float*)d_out;

  // workspace: WTq|WTk|WTv (contiguous, H*E bf16 each), Q,K (B*T*H bf16),
  // VT (B*H*T bf16) => ~12.8 MB total
  __bf16* WTq = (__bf16*)d_ws;
  __bf16* WTk = WTq + (size_t)HH * EE;
  __bf16* WTv = WTk + (size_t)HH * EE;
  __bf16* Qb  = WTv + (size_t)HH * EE;
  __bf16* Kb  = Qb + (size_t)BB * TT * HH;
  __bf16* VT  = Kb + (size_t)BB * TT * HH;

  wt_kernel<<<(HH * EE + 255) / 256, 256, 0, stream>>>(Wk, Wq, Wv, WTq, WTk, WTv);
  qkv_proj_kernel<<<BB * 4, 256, 0, stream>>>(x, WTq, Qb, Kb, VT);
  attn_kernel<<<(BB * TT / 16) / 8, 256, 0, stream>>>(Qb, Kb, VT, amask, out);
}
